// Transformer_26010321945301
// MI455X (gfx1250) — matrix operand order repaired and checked by reading, not compiled, not run on hardware
//
#include <hip/hip_runtime.h>
#include <hip/hip_bf16.h>

// ---------------------------------------------------------------------------
// Transformer forward for MI455X (gfx1250, wave32, WMMA 16x16x32 bf16).
// B=2, N=M=1024, DIM=INNER=1024, HEADS=16, DH=64, MLP=4096, DEPTH=2.
// ~172 GFLOP, weights (64MB bf16 packed) are L2-resident (192MB) ->
// compute-bound: every GEMM / attention einsum runs on
// v_wmma_f32_16x16x32_bf16 with f32 accumulation.
// All hot-loop addressing is base-pointer(+int offset) off __restrict args so
// clang's addrspace inference keeps GLOBAL_LOAD_B128 (not FLAT) in the loops.
// Softmax row reductions use v_permlane16_b32 (VALU) instead of ds_bpermute.
// ---------------------------------------------------------------------------

#define TB 2
#define TN 1024
#define TM 1024
#define TDIM 1024
#define TINNER 1024
#define THEADS 16
#define TDH 64
#define TMLP 4096
#define TDEPTH 2
#define TROWS (TB * TN)          // 2048
#define TSCALE 0.125f            // 64^-0.5

typedef __attribute__((ext_vector_type(16))) __bf16 bf16x16;
typedef __attribute__((ext_vector_type(8)))  __bf16 bf16x8;
typedef __attribute__((ext_vector_type(8)))  float  f32x8;

union FragA {            // 16x32 bf16 A operand, 16 elems / lane
    bf16x16 v;
    bf16x8  h[2];
};

__device__ __forceinline__ f32x8 wmma_bf16(bf16x16 a, bf16x16 b, f32x8 c) {
    return __builtin_amdgcn_wmma_f32_16x16x32_bf16(
        /*neg_a=*/false, a, /*neg_b=*/false, b,
        /*c_mod=*/(short)0, c, /*reuse_a=*/false, /*reuse_b=*/false);
}

// Intra-16-lane xor butterfly reductions. v_permlane16_b32 keeps these in the
// VALU (lane-select tables encode perm[i]=i^mask); falls back to shfl_xor
// (ds_bpermute) if the builtin is unavailable.
#if __has_builtin(__builtin_amdgcn_permlane16)
#define LANE16_XOR(x, lo, hi)                                                \
    __uint_as_float(__builtin_amdgcn_permlane16(                             \
        __float_as_uint(x), __float_as_uint(x), (lo), (hi), false, false))
__device__ __forceinline__ float red_max16(float x) {
    x = fmaxf(x, LANE16_XOR(x, 0xFEDCBA98u, 0x76543210u));   // ^8
    x = fmaxf(x, LANE16_XOR(x, 0x32107654u, 0xBA98FEDCu));   // ^4
    x = fmaxf(x, LANE16_XOR(x, 0x54761032u, 0xDCFE98BAu));   // ^2
    x = fmaxf(x, LANE16_XOR(x, 0x67452301u, 0xEFCDAB89u));   // ^1
    return x;
}
__device__ __forceinline__ float red_add16(float x) {
    x += LANE16_XOR(x, 0xFEDCBA98u, 0x76543210u);
    x += LANE16_XOR(x, 0x32107654u, 0xBA98FEDCu);
    x += LANE16_XOR(x, 0x54761032u, 0xDCFE98BAu);
    x += LANE16_XOR(x, 0x67452301u, 0xEFCDAB89u);
    return x;
}
#else
__device__ __forceinline__ float red_max16(float x) {
#pragma unroll
    for (int m = 8; m >= 1; m >>= 1) x = fmaxf(x, __shfl_xor(x, m, 32));
    return x;
}
__device__ __forceinline__ float red_add16(float x) {
#pragma unroll
    for (int m = 8; m >= 1; m >>= 1) x += __shfl_xor(x, m, 32);
    return x;
}
#endif

// ---------------------------------------------------------------------------
// Weight pack: f32 [K,N] row-major -> bf16 B-fragment order.
// 32x16 tile t=(k/32)*(N/16)+(n/16); lane=(n&15)+16*((k>>4)&1); elem j=k&15.
// GEMM lanes then read their whole fragment as one contiguous 32B chunk.
// ---------------------------------------------------------------------------
__global__ __launch_bounds__(256)
void pack_b_kernel(const float* __restrict__ W, __bf16* __restrict__ out,
                   int K, int N) {
    int idx = blockIdx.x * 256 + threadIdx.x;
    int k = idx / N, n = idx % N;
    int tile = (k >> 5) * (N >> 4) + (n >> 4);
    int lane = (n & 15) + (((k >> 4) & 1) << 4);
    int j = k & 15;
    out[(size_t)tile * 512 + lane * 16 + j] = (__bf16)W[idx];
}

__global__ __launch_bounds__(256)
void cvt_bf16_kernel(const float* __restrict__ in, __bf16* __restrict__ out) {
    int i = blockIdx.x * 256 + threadIdx.x;
    out[i] = (__bf16)in[i];
}

// Per-head V transpose: [b, n, h*64+d] (row stride `stride`) -> [b,h,d,n]
__global__ __launch_bounds__(256)
void vtrans_kernel(const __bf16* __restrict__ V, __bf16* __restrict__ Vt,
                   int stride, int Nk) {
    int idx = blockIdx.x * 256 + threadIdx.x;
    int n = idx & (Nk - 1);
    int d = (idx / Nk) & (TDH - 1);
    int h = (idx / (Nk * TDH)) & (THEADS - 1);
    int b = idx / (Nk * TDH * THEADS);
    Vt[idx] = V[(size_t)(b * Nk + n) * stride + h * TDH + d];
}

// ---------------------------------------------------------------------------
// LayerNorm (f32 in) -> bf16 out. One 256-thread block per row of 1024.
// ---------------------------------------------------------------------------
__global__ __launch_bounds__(256)
void ln_bf16_kernel(const float* __restrict__ x, const float* __restrict__ g,
                    const float* __restrict__ bb, __bf16* __restrict__ out) {
    __shared__ float red[16];
    const int row = blockIdx.x;
    const float4 v = ((const float4*)(x + (size_t)row * TDIM))[threadIdx.x];
    float s = v.x + v.y + v.z + v.w;
    float q = v.x * v.x + v.y * v.y + v.z * v.z + v.w * v.w;
#pragma unroll
    for (int m = 16; m >= 1; m >>= 1) {
        s += __shfl_xor(s, m, 32);
        q += __shfl_xor(q, m, 32);
    }
    const int wave = threadIdx.x >> 5, lane = threadIdx.x & 31;
    if (lane == 0) { red[wave] = s; red[8 + wave] = q; }
    __syncthreads();
    if (threadIdx.x == 0) {
        float ts = 0.f, tq = 0.f;
#pragma unroll
        for (int i = 0; i < 8; ++i) { ts += red[i]; tq += red[8 + i]; }
        float mean = ts * (1.f / TDIM);
        float var = tq * (1.f / TDIM) - mean * mean;
        red[0] = mean;
        red[1] = rsqrtf(var + 1e-5f);
    }
    __syncthreads();
    const float mean = red[0], rstd = red[1];
    const int c = threadIdx.x * 4;
    const float4 gv = ((const float4*)g)[threadIdx.x];
    const float4 bv = ((const float4*)bb)[threadIdx.x];
    __bf16* orow = out + (size_t)row * TDIM;
    orow[c + 0] = (__bf16)((v.x - mean) * rstd * gv.x + bv.x);
    orow[c + 1] = (__bf16)((v.y - mean) * rstd * gv.y + bv.y);
    orow[c + 2] = (__bf16)((v.z - mean) * rstd * gv.z + bv.z);
    orow[c + 3] = (__bf16)((v.w - mean) * rstd * gv.w + bv.w);
}

// ---------------------------------------------------------------------------
// GEMM: C[M,N] = A[M,K] (bf16 row-major) x Bp (bf16 fragment-packed).
// 8 waves / block as 4x2, block tile 128x128, wave tile 32x64 (2x4 accs,
// 8 WMMAs per k-step, k-loop unrolled 2x). Weight stream prefetched ahead.
// MODE 0: bf16 out            (qkv / q2 / k2 / v2 projections, no bias)
// MODE 1: f32  out = acc + bias + residual   (o1 / o2 / ff2 + residual x)
// MODE 2: bf16 out = gelu(acc + bias)        (ff1, exact erf GELU)
// ---------------------------------------------------------------------------
template <int MODE>
__global__ __launch_bounds__(256)
void gemm_bf16_kernel(const __bf16* __restrict__ A, const __bf16* __restrict__ Bp,
                      const float* __restrict__ bias, const float* __restrict__ res,
                      float* __restrict__ outF, __bf16* __restrict__ outB,
                      int Mrows, int N, int K) {
    const int lane = threadIdx.x & 31;
    const int wave = threadIdx.x >> 5;
    const int half = lane >> 4, lm = lane & 15;
    const int bm = blockIdx.y * 128 + (wave >> 1) * 32;
    const int bn = blockIdx.x * 128 + (wave & 1) * 64;
    const int ntiles = N >> 4;
    const size_t bstride = (size_t)ntiles * 512;   // packed elems per k-step

    // base pointers derived directly from __restrict args (global addrspace)
    const __bf16* a0 = A + (size_t)(bm + lm) * K + 16 * half;
    const __bf16* a1 = a0 + (size_t)16 * K;
    const __bf16* b0 = Bp + (size_t)(bn >> 4) * 512 + lane * 16;

    f32x8 acc[2][4] = {};
    size_t ao = 0, bo = 0;
#pragma unroll 2
    for (int k0 = 0; k0 < K; k0 += 32) {
        FragA a[2];
        a[0].h[0] = *(const bf16x8*)(a0 + ao);
        a[0].h[1] = *(const bf16x8*)(a0 + ao + 8);
        a[1].h[0] = *(const bf16x8*)(a1 + ao);
        a[1].h[1] = *(const bf16x8*)(a1 + ao + 8);
        bf16x16 b[4];
#pragma unroll
        for (int c = 0; c < 4; ++c) {
            b[c] = *(const bf16x16*)(b0 + bo + (size_t)c * 512);
            __builtin_prefetch((const void*)(b0 + bo + bstride + (size_t)c * 512),
                               0, 1);
        }
        ao += 32;
        bo += bstride;
#pragma unroll
        for (int t = 0; t < 2; ++t)
#pragma unroll
            for (int c = 0; c < 4; ++c)
                acc[t][c] = wmma_bf16(a[t].v, b[c], acc[t][c]);
    }
#pragma unroll
    for (int t = 0; t < 2; ++t)
#pragma unroll
        for (int c = 0; c < 4; ++c) {
#pragma unroll
            for (int r = 0; r < 8; ++r) {
                const int row = bm + t * 16 + r + 8 * half;
                const int col = bn + c * 16 + lm;
                float v = acc[t][c][r];
                if (MODE == 1) {
                    outF[(size_t)row * N + col] =
                        v + bias[col] + res[(size_t)row * N + col];
                } else if (MODE == 2) {
                    v += bias[col];
                    v = 0.5f * v * (1.0f + erff(v * 0.70710678118654752f));
                    outB[(size_t)row * N + col] = (__bf16)v;
                } else {
                    outB[(size_t)row * N + col] = (__bf16)v;
                }
            }
        }
}

// ---------------------------------------------------------------------------
// Flash attention. grid = (Nq/64, B*H), 128 threads = 4 waves, 16 queries/wave.
// Q: A operand (two 16x32 frags over DH=64). K row-major (keys x DH) is the B
// operand with contiguous per-lane 32B frags. S comes back in WMMA C-layout;
// row softmax stats reduced with v_permlane16 butterflies in the 16-lane
// groups. P bounces through a 1KB *per-wave-private* LDS tile (C->A
// re-layout): DS ops within a wave are processed in order (ISA 7.3), so only
// a compiler scheduling fence (wave_barrier, zero instructions) is needed.
// V pre-transposed per head -> contiguous B frags.
// ---------------------------------------------------------------------------
__global__ __launch_bounds__(128)
void attn_kernel(const __bf16* __restrict__ Q, const __bf16* __restrict__ Kmat,
                 const __bf16* __restrict__ Vt, __bf16* __restrict__ Out,
                 int qStride, int kStride, int Nq, int Nk) {
    __shared__ __bf16 lds[4][16][32];
    const int lane = threadIdx.x & 31;
    const int wave = threadIdx.x >> 5;
    const int half = lane >> 4, lm = lane & 15;
    const int b = blockIdx.y >> 4, h = blockIdx.y & 15;

    const __bf16* qb = Q + (size_t)b * Nq * qStride + h * TDH;
    const __bf16* kb = Kmat + (size_t)b * Nk * kStride + h * TDH;
    const __bf16* vb = Vt + (size_t)(b * THEADS + h) * TDH * Nk;
    __bf16* ob = Out + (size_t)b * Nq * TINNER + h * TDH;
    const int qrow = blockIdx.x * 64 + wave * 16;

    FragA qa[2];
#pragma unroll
    for (int t = 0; t < 2; ++t) {
        const __bf16* qp = qb + (size_t)(qrow + lm) * qStride + t * 32 + 16 * half;
        qa[t].h[0] = *(const bf16x8*)(qp);
        qa[t].h[1] = *(const bf16x8*)(qp + 8);
    }

    // lane-fixed bases + running integer offsets (keeps loads GLOBAL)
    const __bf16* kBase = kb + (size_t)lm * kStride + 16 * half;
    const __bf16* vBase = vb + (size_t)lm * Nk + 16 * half;
    const size_t kTile = (size_t)16 * kStride;   // key-tile stride
    const size_t vTile = (size_t)16 * Nk;        // dim-tile stride
    const size_t kChunk = (size_t)32 * kStride;  // per-iteration advance

    float mrow[8], lrow[8];
    f32x8 o[4] = {};
#pragma unroll
    for (int r = 0; r < 8; ++r) { mrow[r] = -1e30f; lrow[r] = 0.f; }

    size_t ko = 0, vo = 0;
    for (int kc = 0; kc < Nk; kc += 32) {
        bf16x16 kf[2][2];
#pragma unroll
        for (int nt = 0; nt < 2; ++nt) {
            kf[nt][0] = *(const bf16x16*)(kBase + ko + nt * kTile);
            kf[nt][1] = *(const bf16x16*)(kBase + ko + nt * kTile + 32);
        }
        ko += kChunk;

        f32x8 st[2];
#pragma unroll
        for (int nt = 0; nt < 2; ++nt) {
            f32x8 z = {};
            z = wmma_bf16(qa[0].v, kf[nt][0], z);
            z = wmma_bf16(qa[1].v, kf[nt][1], z);
            st[nt] = z;
        }

        float p0[8], p1[8], corr[8];
#pragma unroll
        for (int r = 0; r < 8; ++r) {
            float s0 = st[0][r] * TSCALE;
            float s1 = st[1][r] * TSCALE;
            float mx = red_max16(fmaxf(s0, s1));
            float nm = fmaxf(mrow[r], mx);
            corr[r] = __expf(mrow[r] - nm);
            mrow[r] = nm;
            p0[r] = __expf(s0 - nm);
            p1[r] = __expf(s1 - nm);
            float rs = red_add16(p0[r] + p1[r]);
            lrow[r] = lrow[r] * corr[r] + rs;
        }
#pragma unroll
        for (int d = 0; d < 4; ++d)
#pragma unroll
            for (int r = 0; r < 8; ++r) o[d][r] *= corr[r];

        // C-layout -> A-layout via per-wave-private LDS tile
#pragma unroll
        for (int r = 0; r < 8; ++r) {
            lds[wave][r + 8 * half][lm]      = (__bf16)p0[r];
            lds[wave][r + 8 * half][16 + lm] = (__bf16)p1[r];
        }
        __builtin_amdgcn_wave_barrier();
        FragA pa;
        {
            const __bf16* pp = &lds[wave][lm][16 * half];
            pa.h[0] = *(const bf16x8*)(pp);
            pa.h[1] = *(const bf16x8*)(pp + 8);
        }
        __builtin_amdgcn_wave_barrier();

#pragma unroll
        for (int d = 0; d < 4; ++d) {
            bf16x16 vf = *(const bf16x16*)(vBase + vo + d * vTile);
            o[d] = wmma_bf16(pa.v, vf, o[d]);
        }
        vo += 32;
    }

#pragma unroll
    for (int d = 0; d < 4; ++d)
#pragma unroll
        for (int r = 0; r < 8; ++r) {
            float v = o[d][r] / lrow[r];
            ob[(size_t)(qrow + r + 8 * half) * TINNER + d * 16 + lm] = (__bf16)v;
        }
}

// ---------------------------------------------------------------------------
// Host sequencing
// ---------------------------------------------------------------------------
extern "C" void kernel_launch(void* const* d_in, const int* in_sizes, int n_in,
                              void* d_out, int out_size, void* d_ws, size_t ws_size,
                              hipStream_t stream) {
    (void)in_sizes; (void)n_in; (void)out_size; (void)ws_size;

    const float* in_x  = (const float*)d_in[0];
    const float* in_cx = (const float*)d_in[1];
    const float* ln1_g = (const float*)d_in[2];
    const float* ln1_b = (const float*)d_in[3];
    const float* w_qkv = (const float*)d_in[4];
    const float* w_o1  = (const float*)d_in[5];
    const float* b_o1  = (const float*)d_in[6];
    const float* ln2_g = (const float*)d_in[7];
    const float* ln2_b = (const float*)d_in[8];
    const float* w_q   = (const float*)d_in[9];
    const float* w_k   = (const float*)d_in[10];
    const float* w_v   = (const float*)d_in[11];
    const float* w_o2  = (const float*)d_in[12];
    const float* b_o2  = (const float*)d_in[13];
    const float* ln3_g = (const float*)d_in[14];
    const float* ln3_b = (const float*)d_in[15];
    const float* w_ff1 = (const float*)d_in[16];
    const float* b_ff1 = (const float*)d_in[17];
    const float* w_ff2 = (const float*)d_in[18];
    const float* b_ff2 = (const float*)d_in[19];

    char* ws = (char*)d_ws;
    size_t off = 0;
    auto alloc = [&](size_t bytes) -> void* {
        void* p = ws + off;
        off = (off + bytes + 255) & ~(size_t)255;
        return p;
    };

    float*  X    = (float*) alloc((size_t)TROWS * TDIM * 4);
    __bf16* HBF  = (__bf16*)alloc((size_t)TROWS * TDIM * 2);
    __bf16* CTX  = (__bf16*)alloc((size_t)TB * TM * TDIM * 2);
    __bf16* QKV  = (__bf16*)alloc((size_t)TROWS * 3 * TINNER * 2);
    __bf16* Q2   = (__bf16*)alloc((size_t)TROWS * TINNER * 2);
    __bf16* K2   = (__bf16*)alloc((size_t)TB * TM * TINNER * 2);
    __bf16* V2   = (__bf16*)alloc((size_t)TB * TM * TINNER * 2);
    __bf16* VT   = (__bf16*)alloc((size_t)TB * THEADS * TDH * TM * 2);
    __bf16* ATT  = (__bf16*)alloc((size_t)TROWS * TINNER * 2);
    __bf16* FF   = (__bf16*)alloc((size_t)TROWS * TMLP * 2);

    __bf16 *PQKV[TDEPTH], *PO1[TDEPTH], *PQ[TDEPTH], *PK[TDEPTH],
           *PV[TDEPTH], *PO2[TDEPTH], *PF1[TDEPTH], *PF2[TDEPTH];
    for (int l = 0; l < TDEPTH; ++l) {
        PQKV[l] = (__bf16*)alloc((size_t)TDIM * 3 * TINNER * 2);
        PO1[l]  = (__bf16*)alloc((size_t)TINNER * TDIM * 2);
        PQ[l]   = (__bf16*)alloc((size_t)TDIM * TINNER * 2);
        PK[l]   = (__bf16*)alloc((size_t)TDIM * TINNER * 2);
        PV[l]   = (__bf16*)alloc((size_t)TDIM * TINNER * 2);
        PO2[l]  = (__bf16*)alloc((size_t)TINNER * TDIM * 2);
        PF1[l]  = (__bf16*)alloc((size_t)TDIM * TMLP * 2);
        PF2[l]  = (__bf16*)alloc((size_t)TMLP * TDIM * 2);
    }

    auto pack = [&](const float* W, __bf16* dst, int K, int N) {
        pack_b_kernel<<<(K * N) / 256, 256, 0, stream>>>(W, dst, K, N);
    };
    for (int l = 0; l < TDEPTH; ++l) {
        pack(w_qkv + (size_t)l * TDIM * 3 * TINNER, PQKV[l], TDIM, 3 * TINNER);
        pack(w_o1  + (size_t)l * TINNER * TDIM,     PO1[l],  TINNER, TDIM);
        pack(w_q   + (size_t)l * TDIM * TINNER,     PQ[l],   TDIM, TINNER);
        pack(w_k   + (size_t)l * TDIM * TINNER,     PK[l],   TDIM, TINNER);
        pack(w_v   + (size_t)l * TDIM * TINNER,     PV[l],   TDIM, TINNER);
        pack(w_o2  + (size_t)l * TINNER * TDIM,     PO2[l],  TINNER, TDIM);
        pack(w_ff1 + (size_t)l * TDIM * TMLP,       PF1[l],  TDIM, TMLP);
        pack(w_ff2 + (size_t)l * TMLP * TDIM,       PF2[l],  TMLP, TDIM);
    }

    // context -> bf16 (reference does NOT layernorm the context)
    cvt_bf16_kernel<<<(TB * TM * TDIM) / 256, 256, 0, stream>>>(in_cx, CTX);
    // x (f32 residual stream) into workspace
    hipMemcpyAsync(X, in_x, (size_t)TROWS * TDIM * 4,
                   hipMemcpyDeviceToDevice, stream);

    const dim3 blk256(256), blk128(128);
    const dim3 gDxD(TDIM / 128, TROWS / 128);        // (8,16)
    const dim3 gQKV(3 * TINNER / 128, TROWS / 128);  // (24,16)
    const dim3 gFF1(TMLP / 128, TROWS / 128);        // (32,16)
    const dim3 gAtt(TN / 64, TB * THEADS);           // (16,32)

    for (int l = 0; l < TDEPTH; ++l) {
        // ---- self-attention block ----
        ln_bf16_kernel<<<TROWS, blk256, 0, stream>>>(
            X, ln1_g + l * TDIM, ln1_b + l * TDIM, HBF);
        gemm_bf16_kernel<0><<<gQKV, blk256, 0, stream>>>(
            HBF, PQKV[l], nullptr, nullptr, nullptr, QKV,
            TROWS, 3 * TINNER, TDIM);
        vtrans_kernel<<<(TB * THEADS * TDH * TN) / 256, blk256, 0, stream>>>(
            QKV + 2 * TINNER, VT, 3 * TINNER, TN);
        attn_kernel<<<gAtt, blk128, 0, stream>>>(
            QKV, QKV + TINNER, VT, ATT, 3 * TINNER, 3 * TINNER, TN, TN);
        gemm_bf16_kernel<1><<<gDxD, blk256, 0, stream>>>(
            ATT, PO1[l], b_o1 + l * TDIM, X, X, nullptr,
            TROWS, TDIM, TINNER);

        // ---- cross-attention block ----
        ln_bf16_kernel<<<TROWS, blk256, 0, stream>>>(
            X, ln2_g + l * TDIM, ln2_b + l * TDIM, HBF);
        gemm_bf16_kernel<0><<<gDxD, blk256, 0, stream>>>(
            HBF, PQ[l], nullptr, nullptr, nullptr, Q2, TROWS, TINNER, TDIM);
        gemm_bf16_kernel<0><<<gDxD, blk256, 0, stream>>>(
            CTX, PK[l], nullptr, nullptr, nullptr, K2, TB * TM, TINNER, TDIM);
        gemm_bf16_kernel<0><<<gDxD, blk256, 0, stream>>>(
            CTX, PV[l], nullptr, nullptr, nullptr, V2, TB * TM, TINNER, TDIM);
        vtrans_kernel<<<(TB * THEADS * TDH * TM) / 256, blk256, 0, stream>>>(
            V2, VT, TINNER, TM);
        attn_kernel<<<gAtt, blk128, 0, stream>>>(
            Q2, K2, VT, ATT, TINNER, TINNER, TN, TM);
        gemm_bf16_kernel<1><<<gDxD, blk256, 0, stream>>>(
            ATT, PO2[l], b_o2 + l * TDIM, X, X, nullptr,
            TROWS, TDIM, TINNER);

        // ---- FFN block ----
        ln_bf16_kernel<<<TROWS, blk256, 0, stream>>>(
            X, ln3_g + l * TDIM, ln3_b + l * TDIM, HBF);
        gemm_bf16_kernel<2><<<gFF1, blk256, 0, stream>>>(
            HBF, PF1[l], b_ff1 + l * TMLP, nullptr, nullptr, FF,
            TROWS, TMLP, TDIM);
        gemm_bf16_kernel<1><<<gDxD, blk256, 0, stream>>>(
            FF, PF2[l], b_ff2 + l * TDIM, X, X, nullptr,
            TROWS, TDIM, TMLP);
    }

    hipMemcpyAsync(d_out, X, (size_t)TROWS * TDIM * 4,
                   hipMemcpyDeviceToDevice, stream);
}
